// DiffusionTransformerBlock_64252710748743
// MI455X (gfx1250) — matrix operand order repaired and checked by reading, not compiled, not run on hardware
//
#include <hip/hip_runtime.h>
#include <hip/hip_bf16.h>

typedef _Float16 h16;
typedef __attribute__((ext_vector_type(16))) _Float16 v16h;
typedef __attribute__((ext_vector_type(8)))  _Float16 v8h;
typedef __attribute__((ext_vector_type(8)))  float    v8f;
typedef __attribute__((ext_vector_type(4)))  float    v4f;
typedef __attribute__((ext_vector_type(4)))  int      i4;

#define WG 256

// CDNA5 async Global->LDS staging (ASYNCcnt-tracked), if the toolchain has it.
#if defined(__has_builtin)
#  if __has_builtin(__builtin_amdgcn_global_load_async_to_lds_b128) && \
      __has_builtin(__builtin_amdgcn_s_wait_asynccnt)
#    define USE_ASYNC_LDS 1
#  endif
#endif
#ifndef USE_ASYNC_LDS
#  define USE_ASYNC_LDS 0
#endif

#if USE_ASYNC_LDS
typedef __attribute__((address_space(1))) i4 GI4;   // global int4 (prints as __device__)
typedef __attribute__((address_space(3))) i4 LI4;   // LDS int4    (prints as __shared__)
#endif

// ---------------------------------------------------------------------------
// Tiled WMMA GEMM: C[M,N] = A[M,K](f16) @ B[K,N](f16) (+bias), f32 out.
// Preconditions (guaranteed by all call sites):
//   M % 128 == 0, K % 32 == 0, all A/B rows 16B-alignable,
//   B may be over-read by up to 16 columns (stays inside workspace).
// Block = 256 threads = 8 wave32 waves. Block tile 128x64, K-step 32.
// Wave w: wm = w&3 (32 rows), wn = w>>2 (32 cols) -> 2x2 of 16x16 WMMA tiles.
// ---------------------------------------------------------------------------
__global__ __launch_bounds__(WG) void k_gemm(
    const h16* __restrict__ A, int lda,
    const h16* __restrict__ B, int ldb,
    const float* __restrict__ bias,
    float* __restrict__ C, int ldc,
    int M, int Nn, int K)
{
  __shared__ h16 As[128][40];  // 128x32 tile, row stride 80B (16B multiple)
  __shared__ h16 Bs[32][72];   // 32x64 tile,  row stride 144B (16B multiple)

  const int tid  = threadIdx.x;
  const int wave = tid >> 5;
  const int lane = tid & 31;
  const int wm   = wave & 3;
  const int wn   = wave >> 2;
  const int m0   = blockIdx.x * 128;
  const int n0   = blockIdx.y * 64;

  v8f c00 = {}, c01 = {}, c10 = {}, c11 = {};

  const int r  = lane & 15;
  const int kb = (lane >> 4) * 8;

  const int arow = tid >> 1, acol = (tid & 1) * 16;   // A: 2 x 16B per thread
  const int brow = tid >> 3, bcol = (tid & 7) * 8;    // B: 1 x 16B per thread

  const h16* gpA = A + (long)(m0 + arow) * lda + acol;
  const h16* gpB = B + (long)brow * ldb + n0 + bcol;

  for (int k0 = 0; k0 < K; k0 += 32) {
    // ---- stage tiles into LDS ----
#if USE_ASYNC_LDS
    __builtin_amdgcn_global_load_async_to_lds_b128(
        (GI4*)(gpA + k0),              (LI4*)&As[arow][acol],     0, 0);
    __builtin_amdgcn_global_load_async_to_lds_b128(
        (GI4*)(gpA + k0 + 8),          (LI4*)&As[arow][acol + 8], 0, 0);
    __builtin_amdgcn_global_load_async_to_lds_b128(
        (GI4*)(gpB + (long)k0 * ldb),  (LI4*)&Bs[brow][bcol],     0, 0);
#else
    *(v8h*)(&As[arow][acol])     = *(const v8h*)(gpA + k0);
    *(v8h*)(&As[arow][acol + 8]) = *(const v8h*)(gpA + k0 + 8);
    *(v8h*)(&Bs[brow][bcol])     = *(const v8h*)(gpB + (long)k0 * ldb);
#endif
    if (k0 + 32 < K) {
      __builtin_prefetch((const void*)(gpA + k0 + 32), 0, 1);
      __builtin_prefetch((const void*)(gpB + (long)(k0 + 32) * ldb), 0, 1);
    }
#if USE_ASYNC_LDS
    __builtin_amdgcn_s_wait_asynccnt(0);
#endif
    __syncthreads();

    // ---- fragments: contiguous 16B LDS vector loads ----
    const h16* Ap0 = &As[wm * 32 + r][0];
    const h16* Ap1 = &As[wm * 32 + 16 + r][0];
    v8h a0l = *(const v8h*)(Ap0 + kb);
    v8h a0h = *(const v8h*)(Ap0 + kb + 16);
    v8h a1l = *(const v8h*)(Ap1 + kb);
    v8h a1h = *(const v8h*)(Ap1 + kb + 16);
    v16h af0 = __builtin_shufflevector(a0l, a0h, 0,1,2,3,4,5,6,7,8,9,10,11,12,13,14,15);
    v16h af1 = __builtin_shufflevector(a1l, a1h, 0,1,2,3,4,5,6,7,8,9,10,11,12,13,14,15);

    const h16* Bp0 = &Bs[0][wn * 32 + r];
    v16h bf0, bf1;
    #pragma unroll
    for (int s = 0; s < 16; ++s) {
      const int kk = kb + (s < 8 ? s : s + 8);
      bf0[s] = Bp0[kk * 72];
      bf1[s] = Bp0[kk * 72 + 16];
    }

    c00 = __builtin_amdgcn_wmma_f32_16x16x32_f16(false, af0, false, bf0, (short)0, c00, false, false);
    c01 = __builtin_amdgcn_wmma_f32_16x16x32_f16(false, af0, false, bf1, (short)0, c01, false, false);
    c10 = __builtin_amdgcn_wmma_f32_16x16x32_f16(false, af1, false, bf0, (short)0, c10, false, false);
    c11 = __builtin_amdgcn_wmma_f32_16x16x32_f16(false, af1, false, bf1, (short)0, c11, false, false);
    __syncthreads();
  }

  // ---- store C (f32 16x16 accumulator: VGPR j -> M=j+8*(lane>=16), N=lane&15)
  const int rowHi = (lane >> 4) * 8;
  const int col   = lane & 15;
  const int gc0 = n0 + wn * 32 + col;
  const int gc1 = gc0 + 16;
  const float b0 = bias ? bias[gc0 < Nn ? gc0 : 0] : 0.f;
  const float b1 = bias ? bias[gc1 < Nn ? gc1 : 0] : 0.f;
  #pragma unroll
  for (int j = 0; j < 8; ++j) {
    const int gr0 = m0 + wm * 32 + rowHi + j;
    const int gr1 = gr0 + 16;
    if (gc0 < Nn) {
      C[(long)gr0 * ldc + gc0] = c00[j] + b0;
      C[(long)gr1 * ldc + gc0] = c10[j] + b0;
    }
    if (gc1 < Nn) {
      C[(long)gr0 * ldc + gc1] = c01[j] + b1;
      C[(long)gr1 * ldc + gc1] = c11[j] + b1;
    }
  }
}

// ---------------------------------------------------------------------------
// Row LayerNorm (optional gamma/beta), f32 in -> f16 out. One block per row.
// ---------------------------------------------------------------------------
__global__ __launch_bounds__(WG) void k_rowln(
    const float* __restrict__ x, const float* __restrict__ g,
    const float* __restrict__ b, h16* __restrict__ out, int C)
{
  const int row = blockIdx.x;
  const float* xr = x + (long)row * C;
  __shared__ float sm[WG];
  float v[8]; int cnt = 0;
  float s0 = 0.f;
  for (int c = threadIdx.x; c < C; c += WG) { v[cnt] = xr[c]; s0 += v[cnt]; ++cnt; }
  sm[threadIdx.x] = s0; __syncthreads();
  for (int o = WG / 2; o; o >>= 1) {
    if (threadIdx.x < o) sm[threadIdx.x] += sm[threadIdx.x + o];
    __syncthreads();
  }
  const float mu = sm[0] / (float)C; __syncthreads();
  float s1 = 0.f;
  for (int i = 0; i < cnt; ++i) { const float d = v[i] - mu; s1 += d * d; }
  sm[threadIdx.x] = s1; __syncthreads();
  for (int o = WG / 2; o; o >>= 1) {
    if (threadIdx.x < o) sm[threadIdx.x] += sm[threadIdx.x + o];
    __syncthreads();
  }
  const float rs = rsqrtf(sm[0] / (float)C + 1e-5f);
  cnt = 0;
  for (int c = threadIdx.x; c < C; c += WG) {
    float y = (v[cnt++] - mu) * rs;
    if (g) y *= g[c];
    if (b) y += b[c];
    out[(long)row * C + c] = (h16)y;
  }
}

// ---------------------------------------------------------------------------
// Pair bias: one warp per (q,k). Lane i owns 4 contiguous channels (b128 load).
// LN(z[q,k,:]) * zg + zb, dot with wb[:,h], add (+ mask) into logits[h][q][k].
// ---------------------------------------------------------------------------
__global__ __launch_bounds__(WG) void k_pair_bias(
    const float* __restrict__ z, const float* __restrict__ zg,
    const float* __restrict__ zb, const float* __restrict__ wb,
    const int* __restrict__ mask, float* __restrict__ logits, int Nn)
{
  const long warp = (long)blockIdx.x * (WG / 32) + (threadIdx.x >> 5);
  const int lane = threadIdx.x & 31;
  const long total = (long)Nn * Nn;
  if (warp >= total) return;
  const int qi = (int)(warp / Nn);
  const int ki = (int)(warp % Nn);
  const int c0 = lane * 4;

  v4f v = *(const v4f*)(z + ((long)qi * Nn + ki) * 128 + c0);
  v4f gg = *(const v4f*)(zg + c0);
  v4f bb = *(const v4f*)(zb + c0);

  float mu = v[0] + v[1] + v[2] + v[3];
  #pragma unroll
  for (int o = 16; o; o >>= 1) mu += __shfl_xor(mu, o, 32);
  mu *= (1.f / 128.f);
  float var = 0.f;
  #pragma unroll
  for (int i = 0; i < 4; ++i) { const float d = v[i] - mu; var += d * d; }
  #pragma unroll
  for (int o = 16; o; o >>= 1) var += __shfl_xor(var, o, 32);
  const float rs = rsqrtf(var * (1.f / 128.f) + 1e-5f);

  float zn[4];
  #pragma unroll
  for (int i = 0; i < 4; ++i) zn[i] = (v[i] - mu) * rs * gg[i] + bb[i];
  const float mterm = ((float)mask[ki] - 1.f) * 1e9f;

  #pragma unroll
  for (int h = 0; h < 16; ++h) {
    float p = 0.f;
    #pragma unroll
    for (int i = 0; i < 4; ++i) p += zn[i] * wb[(c0 + i) * 16 + h];
    #pragma unroll
    for (int o = 16; o; o >>= 1) p += __shfl_xor(p, o, 32);
    if (lane == 0)
      logits[((long)h * Nn + qi) * Nn + ki] += p + mterm;
  }
}

// ---------------------------------------------------------------------------
// Row softmax, f32 in -> f16 out. One block per (h,q) row.
// ---------------------------------------------------------------------------
__global__ __launch_bounds__(WG) void k_softmax(
    const float* __restrict__ L, h16* __restrict__ A, int C)
{
  const long row = blockIdx.x;
  const float* x = L + row * (long)C;
  __shared__ float sm[WG];
  float v[8]; int cnt = 0;
  float mx = -3.4e38f;
  for (int c = threadIdx.x; c < C; c += WG) { v[cnt] = x[c]; mx = fmaxf(mx, v[cnt]); ++cnt; }
  sm[threadIdx.x] = mx; __syncthreads();
  for (int o = WG / 2; o; o >>= 1) {
    if (threadIdx.x < o) sm[threadIdx.x] = fmaxf(sm[threadIdx.x], sm[threadIdx.x + o]);
    __syncthreads();
  }
  mx = sm[0]; __syncthreads();
  float s = 0.f;
  for (int i = 0; i < cnt; ++i) { v[i] = __expf(v[i] - mx); s += v[i]; }
  sm[threadIdx.x] = s; __syncthreads();
  for (int o = WG / 2; o; o >>= 1) {
    if (threadIdx.x < o) sm[threadIdx.x] += sm[threadIdx.x + o];
    __syncthreads();
  }
  const float inv = 1.f / sm[0];
  cnt = 0;
  for (int c = threadIdx.x; c < C; c += WG)
    A[row * (long)C + c] = (h16)(v[cnt++] * inv);
}

// ---------------------------------------------------------------------------
// Elementwise / packing kernels
// ---------------------------------------------------------------------------
__device__ __forceinline__ float sigf(float x) { return 1.f / (1.f + __expf(-x)); }

__global__ void k_cvt(const float* __restrict__ x, h16* __restrict__ y, long n, float sc) {
  long i = (long)blockIdx.x * WG + threadIdx.x;
  const long st = (long)gridDim.x * WG;
  for (; i < n; i += st) y[i] = (h16)(x[i] * sc);
}

// qp[r][h*64+c] = c<48 ? q[r][h*48+c]*scale : 0   (N rows, H heads)
__global__ void k_pad_q(const float* __restrict__ q, h16* __restrict__ qp,
                        long n, float scale) {
  long i = (long)blockIdx.x * WG + threadIdx.x;
  const long st = (long)gridDim.x * WG;
  for (; i < n; i += st) {
    const int r = (int)(i >> 10);          // / 1024
    const int rem = (int)(i & 1023);
    const int h = rem >> 6, c = rem & 63;
    qp[i] = (c < 48) ? (h16)(q[(long)r * 768 + h * 48 + c] * scale) : (h16)0.f;
  }
}

// kt[h][c][r] = c<48 ? k[r][h*48+c] : 0   (H*64*N elements)
__global__ void k_pad_kt(const float* __restrict__ k, h16* __restrict__ kt, long n) {
  long i = (long)blockIdx.x * WG + threadIdx.x;
  const long st = (long)gridDim.x * WG;
  for (; i < n; i += st) {
    const int r = (int)(i % 768);
    const int hc = (int)(i / 768);
    const int h = hc >> 6, c = hc & 63;
    kt[i] = (c < 48) ? (h16)k[(long)r * 768 + h * 48 + c] : (h16)0.f;
  }
}

__global__ void k_adaln_combine(const float* __restrict__ gp, const float* __restrict__ sk,
                                const h16* __restrict__ alnn, h16* __restrict__ out, long n) {
  long i = (long)blockIdx.x * WG + threadIdx.x;
  const long st = (long)gridDim.x * WG;
  for (; i < n; i += st)
    out[i] = (h16)(sigf(gp[i]) * (float)alnn[i] + sk[i]);
}

__global__ void k_gate_mul(const float* __restrict__ o, const float* __restrict__ gp,
                           h16* __restrict__ out, long n) {
  long i = (long)blockIdx.x * WG + threadIdx.x;
  const long st = (long)gridDim.x * WG;
  for (; i < n; i += st) out[i] = (h16)(o[i] * sigf(gp[i]));
}

__global__ void k_residual(const float* __restrict__ a, const float* __restrict__ gp,
                           const float* __restrict__ ao, float* __restrict__ a1, long n) {
  long i = (long)blockIdx.x * WG + threadIdx.x;
  const long st = (long)gridDim.x * WG;
  for (; i < n; i += st) a1[i] = a[i] + sigf(gp[i]) * ao[i];
}

__global__ void k_swiglu(const float* __restrict__ h1, const float* __restrict__ h2,
                         h16* __restrict__ out, long n) {
  long i = (long)blockIdx.x * WG + threadIdx.x;
  const long st = (long)gridDim.x * WG;
  for (; i < n; i += st) {
    const float x = h1[i];
    out[i] = (h16)((x * sigf(x)) * h2[i]);
  }
}

__global__ void k_final(const float* __restrict__ a1, const float* __restrict__ gp,
                        const float* __restrict__ t, const int* __restrict__ mask,
                        float* __restrict__ out, long n, int C) {
  long i = (long)blockIdx.x * WG + threadIdx.x;
  const long st = (long)gridDim.x * WG;
  for (; i < n; i += st) {
    const float mf = (float)mask[i / C];
    out[i] = a1[i] + sigf(gp[i]) * t[i] * mf;
  }
}

// ---------------------------------------------------------------------------
extern "C" void kernel_launch(void* const* d_in, const int* in_sizes, int n_in,
                              void* d_out, int out_size, void* d_ws, size_t ws_size,
                              hipStream_t stream) {
  const int Nn = 768, CA = 768, CS = 384, Hh = 16, CH = 48, CT = 1536;
  (void)in_sizes; (void)n_in; (void)out_size; (void)ws_size;

  const float* a   = (const float*)d_in[0];
  const float* s   = (const float*)d_in[1];
  const float* z   = (const float*)d_in[2];
  const int*   msk = (const int*)d_in[3];
  const float* attn_s_ln_g     = (const float*)d_in[4];
  const float* attn_ada_gate_w = (const float*)d_in[5];
  const float* attn_ada_gate_b = (const float*)d_in[6];
  const float* attn_ada_skip_w = (const float*)d_in[7];
  const float* wq = (const float*)d_in[8];
  const float* bq = (const float*)d_in[9];
  const float* wk = (const float*)d_in[10];
  const float* wv = (const float*)d_in[11];
  const float* z_ln_g = (const float*)d_in[12];
  const float* z_ln_b = (const float*)d_in[13];
  const float* wb = (const float*)d_in[14];
  const float* wg = (const float*)d_in[15];
  const float* wo = (const float*)d_in[16];
  const float* attn_out_gate_w = (const float*)d_in[17];
  const float* attn_out_gate_b = (const float*)d_in[18];
  const float* tr_s_ln_g     = (const float*)d_in[19];
  const float* tr_ada_gate_w = (const float*)d_in[20];
  const float* tr_ada_gate_b = (const float*)d_in[21];
  const float* tr_ada_skip_w = (const float*)d_in[22];
  const float* tr_w1 = (const float*)d_in[23];
  const float* tr_w2 = (const float*)d_in[24];
  const float* tr_wo = (const float*)d_in[25];
  const float* tr_out_gate_w = (const float*)d_in[26];
  const float* tr_out_gate_b = (const float*)d_in[27];
  float* out = (float*)d_out;

  // ---- workspace carve ----
  char* w = (char*)d_ws;
  size_t off = 0;
  auto alloc = [&](size_t bytes) -> void* {
    void* p = w + off;
    off += (bytes + 255) & ~(size_t)255;
    return p;
  };
  h16* wq16 = (h16*)alloc((size_t)CA * CA * 2);
  h16* wk16 = (h16*)alloc((size_t)CA * CA * 2);
  h16* wv16 = (h16*)alloc((size_t)CA * CA * 2);
  h16* wg16 = (h16*)alloc((size_t)CA * CA * 2);
  h16* wo16 = (h16*)alloc((size_t)CA * CA * 2);
  h16* agw16 = (h16*)alloc((size_t)CS * CA * 2);
  h16* asw16 = (h16*)alloc((size_t)CS * CA * 2);
  h16* aog16 = (h16*)alloc((size_t)CS * CA * 2);
  h16* tgw16 = (h16*)alloc((size_t)CS * CA * 2);
  h16* tsw16 = (h16*)alloc((size_t)CS * CA * 2);
  h16* tog16 = (h16*)alloc((size_t)CS * CA * 2);
  h16* tw1_16 = (h16*)alloc((size_t)CA * CT * 2);
  h16* tw2_16 = (h16*)alloc((size_t)CA * CT * 2);
  h16* two16  = (h16*)alloc((size_t)CT * CA * 2);
  h16* s16    = (h16*)alloc((size_t)Nn * CS * 2);
  h16* slnA16 = (h16*)alloc((size_t)Nn * CS * 2);
  h16* slnT16 = (h16*)alloc((size_t)Nn * CS * 2);
  h16* alnn16 = (h16*)alloc((size_t)Nn * CA * 2);
  h16* aln16  = (h16*)alloc((size_t)Nn * CA * 2);
  h16* qp16 = (h16*)alloc((size_t)Nn * Hh * 64 * 2);   // [N][H*64] padded, scaled
  h16* kt16 = (h16*)alloc((size_t)Hh * 64 * Nn * 2);   // [H][64][N] padded
  h16* v16  = (h16*)alloc((size_t)Nn * CA * 2);
  h16* og16  = (h16*)alloc((size_t)Nn * CA * 2);
  h16* tln16 = (h16*)alloc((size_t)Nn * CA * 2);
  h16* hid16 = (h16*)alloc((size_t)Nn * CT * 2);
  h16* attn16 = (h16*)alloc((size_t)Hh * Nn * Nn * 2);
  float* logits = (float*)alloc((size_t)Hh * Nn * Nn * 4);
  float* t0 = (float*)alloc((size_t)Nn * CA * 4);
  float* t1 = (float*)alloc((size_t)Nn * CA * 4);
  float* t2 = (float*)alloc((size_t)Nn * CA * 4);
  float* gpre = (float*)alloc((size_t)Nn * CA * 4);
  float* ofl  = (float*)alloc((size_t)Nn * CA * 4);
  float* a1   = (float*)alloc((size_t)Nn * CA * 4);
  float* h1f  = (float*)alloc((size_t)Nn * CT * 4);
  float* h2f  = (float*)alloc((size_t)Nn * CT * 4);

  auto cvt = [&](const float* x, h16* y, long n, float sc) {
    int blocks = (int)((n + WG - 1) / WG);
    if (blocks > 4096) blocks = 4096;
    k_cvt<<<blocks, WG, 0, stream>>>(x, y, n, sc);
  };
  auto gemm = [&](const h16* A, int lda, const h16* B, int ldb,
                  const float* bias, float* C, int ldc, int M, int N_, int K) {
    dim3 g((M + 127) / 128, (N_ + 63) / 64);
    k_gemm<<<g, WG, 0, stream>>>(A, lda, B, ldb, bias, C, ldc, M, N_, K);
  };
  auto ew = [&](long n) {
    int blocks = (int)((n + WG - 1) / WG);
    if (blocks > 4096) blocks = 4096;
    return blocks;
  };

  const long nAC = (long)Nn * CA;
  const long nAT = (long)Nn * CT;

  // 0) weight / input conversions f32 -> f16
  cvt(wq, wq16, nAC, 1.f);  cvt(wk, wk16, nAC, 1.f);
  cvt(wv, wv16, nAC, 1.f);  cvt(wg, wg16, nAC, 1.f);
  cvt(wo, wo16, nAC, 1.f);
  cvt(attn_ada_gate_w, agw16, (long)CS * CA, 1.f);
  cvt(attn_ada_skip_w, asw16, (long)CS * CA, 1.f);
  cvt(attn_out_gate_w, aog16, (long)CS * CA, 1.f);
  cvt(tr_ada_gate_w, tgw16, (long)CS * CA, 1.f);
  cvt(tr_ada_skip_w, tsw16, (long)CS * CA, 1.f);
  cvt(tr_out_gate_w, tog16, (long)CS * CA, 1.f);
  cvt(tr_w1, tw1_16, (long)CA * CT, 1.f);
  cvt(tr_w2, tw2_16, (long)CA * CT, 1.f);
  cvt(tr_wo, two16, (long)CT * CA, 1.f);
  cvt(s, s16, (long)Nn * CS, 1.f);

  // 1) AdaLN(attention)
  k_rowln<<<Nn, WG, 0, stream>>>(s, attn_s_ln_g, nullptr, slnA16, CS);
  k_rowln<<<Nn, WG, 0, stream>>>(a, nullptr, nullptr, alnn16, CA);
  gemm(slnA16, CS, agw16, CA, attn_ada_gate_b, t0, CA, Nn, CA, CS);
  gemm(slnA16, CS, asw16, CA, nullptr,          t1, CA, Nn, CA, CS);
  k_adaln_combine<<<ew(nAC), WG, 0, stream>>>(t0, t1, alnn16, aln16, nAC);

  // 2) projections q,k,v,g ; pack q (padded/scaled) and k^T (padded)
  const float qscale = 0.14433756729740643f; // 1/sqrt(48)
  gemm(aln16, CA, wq16, CA, bq,      t0, CA, Nn, CA, CA);
  k_pad_q<<<ew((long)Nn * Hh * 64), WG, 0, stream>>>(t0, qp16, (long)Nn * Hh * 64, qscale);
  gemm(aln16, CA, wk16, CA, nullptr, t1, CA, Nn, CA, CA);
  k_pad_kt<<<ew((long)Hh * 64 * Nn), WG, 0, stream>>>(t1, kt16, (long)Hh * 64 * Nn);
  gemm(aln16, CA, wv16, CA, nullptr, t2, CA, Nn, CA, CA);
  cvt(t2, v16, nAC, 1.f);
  gemm(aln16, CA, wg16, CA, nullptr, gpre, CA, Nn, CA, CA);

  // 3) per-head logits = qp_h[768x64] @ kt_h[64x768]  (no transpose, no edges)
  for (int h = 0; h < Hh; ++h)
    gemm(qp16 + h * 64, Hh * 64, kt16 + (long)h * 64 * Nn, Nn, nullptr,
         logits + (long)h * Nn * Nn, Nn, Nn, Nn, 64);

  // 4) pair bias from z (+ mask), added into logits
  {
    const long warps = (long)Nn * Nn;
    const long blocks = (warps + (WG / 32) - 1) / (WG / 32);
    k_pair_bias<<<(int)blocks, WG, 0, stream>>>(z, z_ln_g, z_ln_b, wb, msk, logits, Nn);
  }

  // 5) softmax rows -> f16 attention
  k_softmax<<<Hh * Nn, WG, 0, stream>>>(logits, attn16, Nn);

  // 6) o_h = attn_h[768x768] @ v_h[768x48]  (B over-read past col 48 is safe)
  for (int h = 0; h < Hh; ++h)
    gemm(attn16 + (long)h * Nn * Nn, Nn, v16 + h * CH, CA, nullptr,
         ofl + h * CH, CA, Nn, CH, Nn);

  // 7) gate, project with wo, outer gate, residual
  k_gate_mul<<<ew(nAC), WG, 0, stream>>>(ofl, gpre, og16, nAC);
  gemm(og16, CA, wo16, CA, nullptr, t0, CA, Nn, CA, CA);
  gemm(s16, CS, aog16, CA, attn_out_gate_b, t1, CA, Nn, CA, CS);
  k_residual<<<ew(nAC), WG, 0, stream>>>(a, t1, t0, a1, nAC);

  // 8) ConditionedTransitionBlock
  k_rowln<<<Nn, WG, 0, stream>>>(s, tr_s_ln_g, nullptr, slnT16, CS);
  k_rowln<<<Nn, WG, 0, stream>>>(a1, nullptr, nullptr, alnn16, CA);
  gemm(slnT16, CS, tgw16, CA, tr_ada_gate_b, t0, CA, Nn, CA, CS);
  gemm(slnT16, CS, tsw16, CA, nullptr,        t1, CA, Nn, CA, CS);
  k_adaln_combine<<<ew(nAC), WG, 0, stream>>>(t0, t1, alnn16, tln16, nAC);
  gemm(tln16, CA, tw1_16, CT, nullptr, h1f, CT, Nn, CT, CA);
  gemm(tln16, CA, tw2_16, CT, nullptr, h2f, CT, Nn, CT, CA);
  k_swiglu<<<ew(nAT), WG, 0, stream>>>(h1f, h2f, hid16, nAT);
  gemm(hid16, CT, two16, CA, nullptr, t0, CA, Nn, CA, CT);
  gemm(s16, CS, tog16, CA, tr_out_gate_b, t1, CA, Nn, CA, CS);
  k_final<<<ew(nAC), WG, 0, stream>>>(a1, t1, t0, msk, out, nAC, CA);
}
